// DSkBart_83528523973021
// MI455X (gfx1250) — matrix operand order repaired and checked by reading, not compiled, not run on hardware
//
#include <hip/hip_runtime.h>
#include <hip/hip_fp16.h>
#include <cstddef>

// ---------------------------------------------------------------------------
// Model dims (compile-time constants from the reference)
// ---------------------------------------------------------------------------
#define VSZ 32000
#define SLEN 64
#define TLEN 64
#define BSZ 64
#define EDIM 128
#define EHD 256
#define DHD 256

typedef __attribute__((ext_vector_type(16))) _Float16 v16h;
typedef __attribute__((ext_vector_type(8)))  _Float16 v8h;
typedef __attribute__((ext_vector_type(8)))  float    v8f;

// ---------------------------------------------------------------------------
// WMMA GEMM:  C[M x N] = A[M x K] * W[N x K]^T  (+ bias[n]) (opt tanh)
// A, W are f16 row-major (K contiguous). C is f32.
// One wave computes a 64(M) x 32(N) tile: 4 M-subtiles x 2 N-subtiles,
// 8 v_wmma per 32-deep K step. B (weight) stream is read exactly once per
// M/64 row-band -> single-pass weight streaming for the vocab GEMM.
// Requires: M%64==0, N%32==0, K%32==0 (true for every GEMM in this model).
// ---------------------------------------------------------------------------
__global__ void s2s_wmma_gemm(const _Float16* __restrict__ A,
                              const _Float16* __restrict__ W,
                              const float* __restrict__ bias,
                              float* __restrict__ C,
                              int K, int lda, int ldb, int ldc, int mode) {
  const int lane = threadIdx.x & 31;
  const int half = lane >> 4;
  const int l16  = lane & 15;
  const int n0   = blockIdx.x * 32;
  const int m0   = blockIdx.y * 64;

  v8f acc[4][2];
#pragma unroll
  for (int mi = 0; mi < 4; ++mi)
#pragma unroll
    for (int ni = 0; ni < 2; ++ni) acc[mi][ni] = v8f{};

  const _Float16* arow[4];
#pragma unroll
  for (int mi = 0; mi < 4; ++mi)
    arow[mi] = A + (size_t)(m0 + mi * 16 + l16) * lda;

  const _Float16* brow0 = W + (size_t)(n0 +  0 + l16) * ldb + half * 16;
  const _Float16* brow1 = W + (size_t)(n0 + 16 + l16) * ldb + half * 16;

  for (int k0 = 0; k0 < K; k0 += 32) {
    // A fragments: 16x32, contiguous layout (lanes 0-15: K 0..15;
    //                                        lanes 16-31: K 16..31)
    v16h a[4];
#pragma unroll
    for (int mi = 0; mi < 4; ++mi) {
      const v8h lo = *(const v8h*)(arow[mi] + k0 + half * 16);
      const v8h hi = *(const v8h*)(arow[mi] + k0 + half * 16 + 8);
#pragma unroll
      for (int i = 0; i < 8; ++i) { a[mi][i] = lo[i]; a[mi][8 + i] = hi[i]; }
    }
    // B fragments: 32x16 each (column n per lane, K range selected by half)
    const v16h b0 = *(const v16h*)(brow0 + k0);
    const v16h b1 = *(const v16h*)(brow1 + k0);
    // Pull the weight stream ahead (global_prefetch_b8)
    __builtin_prefetch(brow0 + k0 + 64, 0, 1);
    __builtin_prefetch(brow1 + k0 + 64, 0, 1);

#pragma unroll
    for (int mi = 0; mi < 4; ++mi) {
      acc[mi][0] = __builtin_amdgcn_wmma_f32_16x16x32_f16(
          false, a[mi], false, b0, (short)0, acc[mi][0], false, false);
      acc[mi][1] = __builtin_amdgcn_wmma_f32_16x16x32_f16(
          false, a[mi], false, b1, (short)0, acc[mi][1], false, false);
    }
  }

  // C/D layout: lanes 0-15 -> M 0..7, lanes 16-31 -> M 8..15 within a tile
#pragma unroll
  for (int ni = 0; ni < 2; ++ni) {
    const int col = n0 + ni * 16 + l16;
    const float bv = bias ? bias[col] : 0.0f;
#pragma unroll
    for (int mi = 0; mi < 4; ++mi) {
      const int row0 = m0 + mi * 16 + half * 8;
#pragma unroll
      for (int r = 0; r < 8; ++r) {
        float v = acc[mi][ni][r] + bv;
        if (mode == 1) v = tanhf(v);
        C[(size_t)(row0 + r) * ldc + col] = v;
      }
    }
  }
}

// ---------------------------------------------------------------------------
// Element-wise helpers
// ---------------------------------------------------------------------------
__global__ void s2s_f32_to_f16(const float* __restrict__ src,
                               _Float16* __restrict__ dst, size_t n) {
  size_t i = (size_t)blockIdx.x * blockDim.x + threadIdx.x;
  if (i < n) dst[i] = (_Float16)src[i];
}

__global__ void s2s_fill0_f32(float* __restrict__ p, size_t n) {
  size_t i = (size_t)blockIdx.x * blockDim.x + threadIdx.x;
  if (i < n) p[i] = 0.0f;
}

__global__ void s2s_gather_f16(const int* __restrict__ tok,
                               const float* __restrict__ table,
                               _Float16* __restrict__ out, int rows, int E) {
  size_t i = (size_t)blockIdx.x * blockDim.x + threadIdx.x;
  if (i >= (size_t)rows * E) return;
  int r = (int)(i / E), c = (int)(i % E);
  out[i] = (_Float16)table[(size_t)tok[r] * E + c];
}

__global__ void s2s_gather_f32(const int* __restrict__ tok,
                               const float* __restrict__ table,
                               float* __restrict__ out, int rows, int E) {
  size_t i = (size_t)blockIdx.x * blockDim.x + threadIdx.x;
  if (i >= (size_t)rows * E) return;
  int r = (int)(i / E), c = (int)(i % E);
  out[i] = table[(size_t)tok[r] * E + c];
}

// Pack up to 3 f32 row-blocks side-by-side into one f16 matrix (GEMM A operand)
__global__ void s2s_pack3_f16(const float* __restrict__ s0, int w0,
                              const float* __restrict__ s1, int w1,
                              const float* __restrict__ s2, int w2,
                              _Float16* __restrict__ dst, int rows) {
  const int W = w0 + w1 + w2;
  size_t i = (size_t)blockIdx.x * blockDim.x + threadIdx.x;
  if (i >= (size_t)rows * W) return;
  int r = (int)(i / W), c = (int)(i % W);
  float v;
  if (c < w0)            v = s0[(size_t)r * w0 + c];
  else if (c < w0 + w1)  v = s1[(size_t)r * w1 + (c - w0)];
  else                   v = s2[(size_t)r * w2 + (c - w0 - w1)];
  dst[i] = (_Float16)v;
}

// GRU gate fusion: gi/gh are [B][3H] (biases already added in GEMM).
// Writes new h (f32 + f16) in place; optionally scatters h into enc_bse.
__global__ void s2s_gru_combine(const float* __restrict__ gi,
                                const float* __restrict__ gh,
                                const float* __restrict__ hprev,
                                float* __restrict__ hout,
                                _Float16* __restrict__ hout16,
                                float* __restrict__ bse, int bse_row_stride,
                                int bse_col_off, int s, int H) {
  size_t i = (size_t)blockIdx.x * blockDim.x + threadIdx.x;
  if (i >= (size_t)BSZ * H) return;
  int b = (int)(i / H), j = (int)(i % H);
  const float* gib = gi + (size_t)b * 3 * H;
  const float* ghb = gh + (size_t)b * 3 * H;
  float r = 1.0f / (1.0f + expf(-(gib[j] + ghb[j])));
  float z = 1.0f / (1.0f + expf(-(gib[H + j] + ghb[H + j])));
  float n = tanhf(gib[2 * H + j] + r * ghb[2 * H + j]);
  float h2 = (1.0f - z) * n + z * hprev[i];
  hout[i] = h2;
  hout16[i] = (_Float16)h2;
  if (bse) bse[(size_t)b * bse_row_stride + (size_t)s * 512 + bse_col_off + j] = h2;
}

// Attention: per batch b compute scores over S, softmax, weighted sum of enc_bse.
__global__ void s2s_attention(const float* __restrict__ enc_proj,  // [B][S][DHD]
                              const float* __restrict__ hW,        // [B][DHD]
                              const float* __restrict__ v,         // [DHD]
                              const float* __restrict__ enc_bse,   // [B][S][512]
                              float* __restrict__ weighted) {      // [B][512]
  __shared__ float sc[SLEN];
  const int b = blockIdx.x;
  const int wave = threadIdx.x >> 5;
  const int lane = threadIdx.x & 31;

  for (int s = wave; s < SLEN; s += 8) {
    const float* ep = enc_proj + ((size_t)b * SLEN + s) * DHD;
    const float* hw = hW + (size_t)b * DHD;
    float acc = 0.0f;
    for (int h = lane; h < DHD; h += 32)
      acc += v[h] * tanhf(ep[h] + hw[h]);
    for (int off = 16; off > 0; off >>= 1)
      acc += __shfl_down(acc, off, 32);
    if (lane == 0) sc[s] = acc;
  }
  __syncthreads();
  if (threadIdx.x == 0) {
    float mx = sc[0];
    for (int s = 1; s < SLEN; ++s) mx = fmaxf(mx, sc[s]);
    float sum = 0.0f;
    for (int s = 0; s < SLEN; ++s) { float e = expf(sc[s] - mx); sc[s] = e; sum += e; }
    float inv = 1.0f / sum;
    for (int s = 0; s < SLEN; ++s) sc[s] *= inv;
  }
  __syncthreads();
  for (int e = threadIdx.x; e < 512; e += blockDim.x) {
    float acc = 0.0f;
    const float* base = enc_bse + (size_t)b * SLEN * 512 + e;
    for (int s = 0; s < SLEN; ++s) acc += sc[s] * base[(size_t)s * 512];
    weighted[(size_t)b * 512 + e] = acc;
  }
}

// ---------------------------------------------------------------------------
// Host orchestration
// ---------------------------------------------------------------------------
static inline size_t s2s_align(size_t x) { return (x + 255) & ~(size_t)255; }

extern "C" void kernel_launch(void* const* d_in, const int* in_sizes, int n_in,
                              void* d_out, int out_size, void* d_ws, size_t ws_size,
                              hipStream_t stream) {
  (void)in_sizes; (void)n_in; (void)out_size; (void)ws_size;

  const int*   src      = (const int*)  d_in[0];   // [S][B]
  const int*   trg      = (const int*)  d_in[1];   // [T][B]
  const float* enc_emb  = (const float*)d_in[2];   // [V][E]
  const float* eWih_f   = (const float*)d_in[3];   // [768][128]
  const float* eWhh_f   = (const float*)d_in[4];   // [768][256]
  const float* ebih_f   = (const float*)d_in[5];
  const float* ebhh_f   = (const float*)d_in[6];
  const float* eWih_b   = (const float*)d_in[7];
  const float* eWhh_b   = (const float*)d_in[8];
  const float* ebih_b   = (const float*)d_in[9];
  const float* ebhh_b   = (const float*)d_in[10];
  const float* efc_W    = (const float*)d_in[11];  // [256][512]
  const float* efc_b    = (const float*)d_in[12];
  const float* attn_W   = (const float*)d_in[13];  // [256][768]
  const float* attn_b   = (const float*)d_in[14];
  const float* attn_v   = (const float*)d_in[15];  // [256]
  const float* dec_emb  = (const float*)d_in[16];  // [V][E]
  const float* dWih     = (const float*)d_in[17];  // [768][640]
  const float* dWhh     = (const float*)d_in[18];  // [768][256]
  const float* dbih     = (const float*)d_in[19];
  const float* dbhh     = (const float*)d_in[20];
  const float* fc_W     = (const float*)d_in[21];  // [32000][896]
  const float* fc_b     = (const float*)d_in[22];
  float* out = (float*)d_out;                      // [T][B][V]

  // ---- workspace carve ----
  char* w = (char*)d_ws; size_t off = 0;
  auto carve = [&](size_t bytes) { void* p = w + off; off = s2s_align(off + bytes); return p; };

  _Float16* fcW16    = (_Float16*)carve((size_t)VSZ * 896 * 2);
  _Float16* dWih16   = (_Float16*)carve((size_t)768 * 640 * 2);
  _Float16* dWhh16   = (_Float16*)carve((size_t)768 * 256 * 2);
  _Float16* eWihF16  = (_Float16*)carve((size_t)768 * 128 * 2);
  _Float16* eWhhF16  = (_Float16*)carve((size_t)768 * 256 * 2);
  _Float16* eWihB16  = (_Float16*)carve((size_t)768 * 128 * 2);
  _Float16* eWhhB16  = (_Float16*)carve((size_t)768 * 256 * 2);
  _Float16* efcW16   = (_Float16*)carve((size_t)256 * 512 * 2);
  _Float16* attnW16  = (_Float16*)carve((size_t)256 * 768 * 2);
  _Float16* emb16    = (_Float16*)carve((size_t)SLEN * BSZ * EDIM * 2);
  float*    enc_bse  = (float*)   carve((size_t)BSZ * SLEN * 512 * 4);
  _Float16* enc_bse16= (_Float16*)carve((size_t)BSZ * SLEN * 512 * 2);
  float*    enc_proj = (float*)   carve((size_t)BSZ * SLEN * DHD * 4);
  float*    hF       = (float*)   carve((size_t)BSZ * DHD * 4);
  _Float16* hF16     = (_Float16*)carve((size_t)BSZ * DHD * 2);
  float*    hB       = (float*)   carve((size_t)BSZ * DHD * 4);
  _Float16* hB16     = (_Float16*)carve((size_t)BSZ * DHD * 2);
  float*    gi       = (float*)   carve((size_t)BSZ * 768 * 4);
  float*    gh       = (float*)   carve((size_t)BSZ * 768 * 4);
  float*    hWbuf    = (float*)   carve((size_t)BSZ * DHD * 4);
  float*    embd     = (float*)   carve((size_t)BSZ * EDIM * 4);
  float*    weighted = (float*)   carve((size_t)BSZ * 512 * 4);
  _Float16* x16      = (_Float16*)carve((size_t)BSZ * 640 * 2);
  _Float16* cat16    = (_Float16*)carve((size_t)BSZ * 896 * 2);
  _Float16* hid16    = (_Float16*)carve((size_t)BSZ * 512 * 2);

  auto conv = [&](const float* s, _Float16* d, size_t n) {
    s2s_f32_to_f16<<<dim3((unsigned)((n + 255) / 256)), 256, 0, stream>>>(s, d, n);
  };
  auto gemm = [&](const _Float16* A, const _Float16* Wm, const float* bias, float* C,
                  int M, int N, int K, int lda, int ldb, int ldc, int mode) {
    s2s_wmma_gemm<<<dim3(N / 32, M / 64), 32, 0, stream>>>(A, Wm, bias, C, K, lda, ldb, ldc, mode);
  };

  // ---- one-time (per launch) weight conversion to f16 ----
  conv(fc_W,   fcW16,   (size_t)VSZ * 896);
  conv(dWih,   dWih16,  (size_t)768 * 640);
  conv(dWhh,   dWhh16,  (size_t)768 * 256);
  conv(eWih_f, eWihF16, (size_t)768 * 128);
  conv(eWhh_f, eWhhF16, (size_t)768 * 256);
  conv(eWih_b, eWihB16, (size_t)768 * 128);
  conv(eWhh_b, eWhhB16, (size_t)768 * 256);
  conv(efc_W,  efcW16,  (size_t)256 * 512);
  conv(attn_W, attnW16, (size_t)256 * 768);

  // ---- output row 0 = zeros ----
  {
    size_t n = (size_t)BSZ * VSZ;
    s2s_fill0_f32<<<dim3((unsigned)((n + 255) / 256)), 256, 0, stream>>>(out, n);
  }

  // ---- encoder embeddings (f16, GEMM A operand) ----
  {
    size_t n = (size_t)SLEN * BSZ * EDIM;
    s2s_gather_f16<<<dim3((unsigned)((n + 255) / 256)), 256, 0, stream>>>(
        src, enc_emb, emb16, SLEN * BSZ, EDIM);
  }

  const size_t BH = (size_t)BSZ * DHD;
  // ---- forward encoder GRU ----
  s2s_fill0_f32<<<dim3((unsigned)((BH + 255) / 256)), 256, 0, stream>>>(hF, BH);
  conv(hF, hF16, BH);
  for (int s = 0; s < SLEN; ++s) {
    gemm(emb16 + (size_t)s * BSZ * EDIM, eWihF16, ebih_f, gi, BSZ, 768, 128, 128, 128, 768, 0);
    gemm(hF16, eWhhF16, ebhh_f, gh, BSZ, 768, 256, 256, 256, 768, 0);
    s2s_gru_combine<<<dim3((unsigned)((BH + 255) / 256)), 256, 0, stream>>>(
        gi, gh, hF, hF, hF16, enc_bse, SLEN * 512, 0, s, EHD);
  }
  // ---- backward encoder GRU ----
  s2s_fill0_f32<<<dim3((unsigned)((BH + 255) / 256)), 256, 0, stream>>>(hB, BH);
  conv(hB, hB16, BH);
  for (int i = 0; i < SLEN; ++i) {
    int s = SLEN - 1 - i;
    gemm(emb16 + (size_t)s * BSZ * EDIM, eWihB16, ebih_b, gi, BSZ, 768, 128, 128, 128, 768, 0);
    gemm(hB16, eWhhB16, ebhh_b, gh, BSZ, 768, 256, 256, 256, 768, 0);
    s2s_gru_combine<<<dim3((unsigned)((BH + 255) / 256)), 256, 0, stream>>>(
        gi, gh, hB, hB, hB16, enc_bse, SLEN * 512, 256, s, EHD);
  }

  // ---- hidden = tanh([hF|hB] @ enc_fc_W.T + b) ----
  {
    size_t n = (size_t)BSZ * 512;
    s2s_pack3_f16<<<dim3((unsigned)((n + 255) / 256)), 256, 0, stream>>>(
        hF, EHD, hB, EHD, (const float*)nullptr, 0, hid16, BSZ);
  }
  gemm(hid16, efcW16, efc_b, hF, BSZ, DHD, 512, 512, 512, DHD, /*tanh*/1);
  conv(hF, hF16, BH);

  // ---- enc_proj = enc_bse @ W_e.T + attn_b  (W_e = attn_W[:, 256:768]) ----
  conv(enc_bse, enc_bse16, (size_t)BSZ * SLEN * 512);
  gemm(enc_bse16, attnW16 + 256, attn_b, enc_proj, BSZ * SLEN, DHD, 512, 512, 768, DHD, 0);

  // ---- decoder ----
  for (int t = 0; t < TLEN - 1; ++t) {
    {
      size_t n = (size_t)BSZ * EDIM;
      s2s_gather_f32<<<dim3((unsigned)((n + 255) / 256)), 256, 0, stream>>>(
          trg + (size_t)t * BSZ, dec_emb, embd, BSZ, EDIM);
    }
    // hW = h @ W_h.T  (W_h = attn_W[:, 0:256])
    gemm(hF16, attnW16, (const float*)nullptr, hWbuf, BSZ, DHD, 256, 256, 768, DHD, 0);
    s2s_attention<<<dim3(BSZ), 256, 0, stream>>>(enc_proj, hWbuf, attn_v, enc_bse, weighted);
    // x = [embd | weighted]  -> f16
    {
      size_t n = (size_t)BSZ * 640;
      s2s_pack3_f16<<<dim3((unsigned)((n + 255) / 256)), 256, 0, stream>>>(
          embd, EDIM, weighted, 512, (const float*)nullptr, 0, x16, BSZ);
    }
    gemm(x16, dWih16, dbih, gi, BSZ, 768, 640, 640, 640, 768, 0);
    gemm(hF16, dWhh16, dbhh, gh, BSZ, 768, 256, 256, 256, 768, 0);
    s2s_gru_combine<<<dim3((unsigned)((BH + 255) / 256)), 256, 0, stream>>>(
        gi, gh, hF, hF, hF16, (float*)nullptr, 0, 0, 0, DHD);
    // pred = [h2 | weighted | embd] @ fc_W.T + fc_b   -> out[t+1]
    {
      size_t n = (size_t)BSZ * 896;
      s2s_pack3_f16<<<dim3((unsigned)((n + 255) / 256)), 256, 0, stream>>>(
          hF, DHD, weighted, 512, embd, EDIM, cat16, BSZ);
    }
    gemm(cat16, fcW16, fc_b, out + (size_t)(t + 1) * BSZ * VSZ,
         BSZ, VSZ, 896, 896, 896, VSZ, 0);
  }
}
